// Transformer_7387343749565
// MI455X (gfx1250) — compile-verified, matrix operand order checked by reading, not run on hardware
//
#include <hip/hip_runtime.h>
#include <math.h>

// ---------------------------------------------------------------------------
// Types for CDNA5 WMMA (wave32): V_WMMA_F32_16X16X32_BF16
// ---------------------------------------------------------------------------
typedef __attribute__((ext_vector_type(16))) __bf16 bf16x16;
typedef __attribute__((ext_vector_type(8)))  float  f32x8;

static __device__ __forceinline__ unsigned int rnd16(unsigned int u) {
  return u + 0x7FFFu + ((u >> 16) & 1u);   // round-to-nearest-even into [31:16]
}
// pack two f32 -> two bf16 in one dword, pure u32 ops (no b16 subreg moves)
static __device__ __forceinline__ unsigned int pk2bf(float a, float b) {
  unsigned int ua = rnd16(__builtin_bit_cast(unsigned int, a)) >> 16;
  unsigned int ub = rnd16(__builtin_bit_cast(unsigned int, b)) & 0xFFFF0000u;
  return ua | ub;
}
static __device__ __forceinline__ __bf16 f2bf(float f) {
  unsigned short s = (unsigned short)(rnd16(__builtin_bit_cast(unsigned int, f)) >> 16);
  return __builtin_bit_cast(__bf16, s);
}

#define LEN_TOT 5440   // 64*64 + 32*32 + 16*16 + 8*8
#define NB      8
#define DMODEL  512
#define NQ_DEC  300

// ---------------------------------------------------------------------------
// WMMA GEMM:  C[M,N] = act( A[M,K] @ W[N,K]^T + bias[N] ), optional row mask.
// ACT: 0=none 1=relu 2=sigmoid
// AMODE 0: A row-major [M,K], output row m. Partial tiles via clamped loads.
// AMODE 1: A is NCHW level source: A(m,k)=A[(b*K+k)*hw + p], b=m/hw, p=m%hw;
//          output row = b*LEN_TOT + s0 + p. (M always a multiple of 128 here.)
// Block: 256 threads = 8 waves; tile 128x64; each wave 32x32 (2x2 WMMA tiles).
// K must be a multiple of 32 (true for all GEMMs: 512 or 2048).
// ---------------------------------------------------------------------------
template<int ACT, int AMODE>
__global__ __launch_bounds__(256) void gemm_wmma(
    const float* __restrict__ A, const float* __restrict__ Wt,
    const float* __restrict__ bias, float* __restrict__ C,
    const unsigned char* __restrict__ rowmask,
    int M, int N, int K, int hw, int s0)
{
  __shared__ __bf16 As[128][40];
  __shared__ __bf16 Bs[64][40];
  const int tid  = threadIdx.x;
  const int wave = tid >> 5, lane = tid & 31;
  const int bm = blockIdx.x * 128, bn = blockIdx.y * 64;
  const int wm = (wave >> 1) * 32, wn = (wave & 1) * 32;
  const int lr = lane & 15, kb = (lane >> 4) * 8;
  f32x8 acc[2][2] = {};

  for (int k0 = 0; k0 < K; k0 += 32) {
    float4 va[4], vb[2];
    // ---- load phase: issue all tile loads, no consumption in between ----
    if (AMODE == 0) {
#pragma unroll
      for (int it = 0; it < 4; it++) {
        int i = tid + it * 256;
        int r = i >> 3, c4 = (i & 7) * 4;
        int m = bm + r;
        int mc = (m < M) ? m : (M - 1);
        va[it] = *(const float4*)(A + (size_t)mc * K + (k0 + c4));
      }
    } else {
#pragma unroll
      for (int it = 0; it < 4; it++) {
        int i = tid + it * 256;
        int c = i & 31, rg = i >> 5;          // rows rg*4 .. rg*4+3
        int m = bm + rg * 4;
        int b = m / hw, p = m - b * hw;
        va[it] = *(const float4*)(A + ((size_t)b * K + (k0 + c)) * hw + p);
      }
    }
#pragma unroll
    for (int it = 0; it < 2; it++) {
      int i = tid + it * 256;
      int r = i >> 3, c4 = (i & 7) * 4;
      int n = bn + r;
      int nc = (n < N) ? n : (N - 1);
      vb[it] = *(const float4*)(Wt + (size_t)nc * K + (k0 + c4));
    }
    // prefetch next K-slab while this one is consumed
    if (AMODE == 0 && k0 + 32 < K) {
      const float* pf;
      if (tid < 128) {
        int m = bm + tid; if (m >= M) m = M - 1;
        pf = A + (size_t)m * K + (k0 + 32);
      } else {
        int n = bn + ((tid - 128) & 63); if (n >= N) n = N - 1;
        pf = Wt + (size_t)n * K + (k0 + 32);
      }
      __builtin_prefetch(pf, 0, 1);
    }
    // ---- pack/store phase ----
    if (AMODE == 0) {
#pragma unroll
      for (int it = 0; it < 4; it++) {
        int i = tid + it * 256;
        int r = i >> 3, c4 = (i & 7) * 4;
        int m = bm + r;
        float4 v = va[it];
        if (m >= M) { v.x = 0.f; v.y = 0.f; v.z = 0.f; v.w = 0.f; }
        unsigned int* dst = (unsigned int*)&As[r][c4];
        dst[0] = pk2bf(v.x, v.y);
        dst[1] = pk2bf(v.z, v.w);
      }
    } else {
#pragma unroll
      for (int it = 0; it < 4; it++) {
        int i = tid + it * 256;
        int c = i & 31, rg = i >> 5;
        float4 v = va[it];
        As[rg * 4 + 0][c] = f2bf(v.x);
        As[rg * 4 + 1][c] = f2bf(v.y);
        As[rg * 4 + 2][c] = f2bf(v.z);
        As[rg * 4 + 3][c] = f2bf(v.w);
      }
    }
#pragma unroll
    for (int it = 0; it < 2; it++) {
      int i = tid + it * 256;
      int r = i >> 3, c4 = (i & 7) * 4;
      int n = bn + r;
      float4 v = vb[it];
      if (n >= N) { v.x = 0.f; v.y = 0.f; v.z = 0.f; v.w = 0.f; }
      unsigned int* dst = (unsigned int*)&Bs[r][c4];
      dst[0] = pk2bf(v.x, v.y);
      dst[1] = pk2bf(v.z, v.w);
    }
    __syncthreads();

    bf16x16 af[2], bfv[2];
#pragma unroll
    for (int mi = 0; mi < 2; mi++) {
      int row = wm + mi * 16 + lr;
#pragma unroll
      for (int e = 0; e < 8; e++) {
        af[mi][e]     = As[row][kb + e];
        af[mi][e + 8] = As[row][16 + kb + e];
      }
    }
#pragma unroll
    for (int ni = 0; ni < 2; ni++) {
      int col = wn + ni * 16 + lr;
#pragma unroll
      for (int e = 0; e < 8; e++) {
        bfv[ni][e]     = Bs[col][kb + e];
        bfv[ni][e + 8] = Bs[col][16 + kb + e];
      }
    }
#pragma unroll
    for (int mi = 0; mi < 2; mi++)
#pragma unroll
      for (int ni = 0; ni < 2; ni++)
        acc[mi][ni] = __builtin_amdgcn_wmma_f32_16x16x32_bf16(
            false, af[mi], false, bfv[ni], (short)0, acc[mi][ni], false, false);
    __syncthreads();
  }

  const int half = lane >> 4;  // C/D layout: lanes 0-15 rows 0-7, lanes 16-31 rows 8-15
#pragma unroll
  for (int mi = 0; mi < 2; mi++)
#pragma unroll
    for (int ni = 0; ni < 2; ni++) {
      int n = bn + wn + ni * 16 + lr;
      if (n >= N) continue;
      float bv = bias ? bias[n] : 0.f;
#pragma unroll
      for (int r = 0; r < 8; r++) {
        int m = bm + wm + mi * 16 + half * 8 + r;
        if (m >= M) continue;
        float v = acc[mi][ni][r] + bv;
        if (ACT == 1) v = fmaxf(v, 0.f);
        if (ACT == 2) v = 1.f / (1.f + __expf(-v));
        if (rowmask && rowmask[m]) v = 0.f;
        size_t orow;
        if (AMODE == 0) orow = (size_t)m;
        else { int b = m / hw, p = m - b * hw; orow = (size_t)b * LEN_TOT + s0 + p; }
        C[orow * (size_t)N + n] = v;
      }
    }
}

// ---------------------------------------------------------------------------
// Elementwise / bookkeeping kernels
// ---------------------------------------------------------------------------
__global__ void add_kernel(const float* __restrict__ a, const float* __restrict__ b,
                           float* __restrict__ o, size_t n)
{
  size_t i = (size_t)blockIdx.x * 256 + threadIdx.x;
  if (i < n) o[i] = a[i] + b[i];
}

// pos_flat[b,s,c] = pos_l[c*HW + p] + level_embed[lvl, c]
__global__ void posflat_kernel(const float* __restrict__ p0, const float* __restrict__ p1,
                               const float* __restrict__ p2, const float* __restrict__ p3,
                               const float* __restrict__ lev, float* __restrict__ posf)
{
  size_t i = (size_t)blockIdx.x * 256 + threadIdx.x;
  if (i >= (size_t)NB * LEN_TOT * DMODEL) return;
  int c = (int)(i % DMODEL);
  size_t r = i / DMODEL;
  int s = (int)(r % LEN_TOT);
  int lvl, hw, p; const float* pp;
  if (s < 4096)      { lvl = 0; hw = 4096; p = s;        pp = p0; }
  else if (s < 5120) { lvl = 1; hw = 1024; p = s - 4096; pp = p1; }
  else if (s < 5376) { lvl = 2; hw = 256;  p = s - 5120; pp = p2; }
  else               { lvl = 3; hw = 64;   p = s - 5376; pp = p3; }
  posf[i] = pp[(size_t)c * hw + p] + lev[lvl * DMODEL + c];
}

__global__ void maskflat_kernel(const unsigned char* __restrict__ m0, const unsigned char* __restrict__ m1,
                                const unsigned char* __restrict__ m2, const unsigned char* __restrict__ m3,
                                unsigned char* __restrict__ mf)
{
  int i = blockIdx.x * 256 + threadIdx.x;
  if (i >= NB * LEN_TOT) return;
  int b = i / LEN_TOT, s = i % LEN_TOT;
  const unsigned char* mp; int p, hw;
  if (s < 4096)      { mp = m0; p = s;        hw = 4096; }
  else if (s < 5120) { mp = m1; p = s - 4096; hw = 1024; }
  else if (s < 5376) { mp = m2; p = s - 5120; hw = 256;  }
  else               { mp = m3; p = s - 5376; hw = 64;   }
  mf[i] = mp[(size_t)b * hw + p];
}

__global__ void vr_kernel(const unsigned char* __restrict__ m0, const unsigned char* __restrict__ m1,
                          const unsigned char* __restrict__ m2, const unsigned char* __restrict__ m3,
                          float* __restrict__ vr)
{
  int i = threadIdx.x;
  if (i >= 32) return;
  int b = i >> 2, l = i & 3;
  const int HS[4] = {64, 32, 16, 8};
  int H = HS[l], W = HS[l];
  const unsigned char* mp = (l == 0) ? m0 : (l == 1) ? m1 : (l == 2) ? m2 : m3;
  int cH = 0, cW = 0;
  for (int h = 0; h < H; h++) cH += mp[((size_t)b * H + h) * W] ? 0 : 1;
  for (int w = 0; w < W; w++) cW += mp[(size_t)b * H * W + w] ? 0 : 1;
  vr[i * 2 + 0] = (float)cW / (float)W;
  vr[i * 2 + 1] = (float)cH / (float)H;
}

// encoder reference points: eref[b,s,l,2]
__global__ void enc_ref_kernel(const float* __restrict__ vr, float* __restrict__ eref)
{
  int i = blockIdx.x * 256 + threadIdx.x;
  if (i >= NB * LEN_TOT) return;
  int b = i / LEN_TOT, s = i % LEN_TOT;
  int lvl, H, W, p;
  if (s < 4096)      { lvl = 0; H = 64; W = 64; p = s;        }
  else if (s < 5120) { lvl = 1; H = 32; W = 32; p = s - 4096; }
  else if (s < 5376) { lvl = 2; H = 16; W = 16; p = s - 5120; }
  else               { lvl = 3; H = 8;  W = 8;  p = s - 5376; }
  int h = p / W, w = p % W;
  float ry = (h + 0.5f) / (vr[(b * 4 + lvl) * 2 + 1] * H);
  float rx = (w + 0.5f) / (vr[(b * 4 + lvl) * 2 + 0] * W);
  for (int l = 0; l < 4; l++) {
    eref[((size_t)i * 4 + l) * 2 + 0] = rx * vr[(b * 4 + l) * 2 + 0];
    eref[((size_t)i * 4 + l) * 2 + 1] = ry * vr[(b * 4 + l) * 2 + 1];
  }
}

// softmax over groups of 16 (attention weights per head)
__global__ void softmax16_kernel(float* __restrict__ aw, size_t ngroups)
{
  size_t i = (size_t)blockIdx.x * 256 + threadIdx.x;
  if (i >= ngroups) return;
  float* p = aw + i * 16;
  float mx = p[0];
  for (int j = 1; j < 16; j++) mx = fmaxf(mx, p[j]);
  float e[16], sm = 0.f;
  for (int j = 0; j < 16; j++) { e[j] = __expf(p[j] - mx); sm += e[j]; }
  float inv = 1.f / sm;
  for (int j = 0; j < 16; j++) p[j] = e[j] * inv;
}

// deformable bilinear sampling: one block per (b,q,head), 64 lanes = dh
__global__ __launch_bounds__(64) void deform_sample_kernel(
    const float* __restrict__ value, const float* __restrict__ so,
    const float* __restrict__ aw, const float* __restrict__ ref,
    float* __restrict__ out, int Lq)
{
  int idx = blockIdx.x;
  int h = idx & 7;
  int t = idx >> 3;
  int q = t % Lq, b = t / Lq;
  int d = threadIdx.x;
  const int HS[4] = {64, 32, 16, 8};
  const int S0[4] = {0, 4096, 5120, 5376};
  size_t row = (size_t)b * Lq + q;
  const float* soP = so + row * 256 + h * 32;
  const float* awP = aw + row * 128 + h * 16;
  const float* rfP = ref + row * 8;
  float acc = 0.f;
  for (int l = 0; l < 4; l++) {
    int H = HS[l], W = HS[l], s0 = S0[l];
    float rx = rfP[l * 2 + 0], ry = rfP[l * 2 + 1];
    for (int p = 0; p < 4; p++) {
      float lx = rx + soP[l * 8 + p * 2 + 0] / (float)W;
      float ly = ry + soP[l * 8 + p * 2 + 1] / (float)H;
      float wgt = awP[l * 4 + p];
      float xx = lx * W - 0.5f, yy = ly * H - 0.5f;
      float x0f = floorf(xx), y0f = floorf(yy);
      float fx = xx - x0f, fy = yy - y0f;
      int x0 = (int)x0f, y0 = (int)y0f;
      float v00 = 0.f, v10 = 0.f, v01 = 0.f, v11 = 0.f;
      if (y0 >= 0 && y0 < H) {
        if (x0 >= 0 && x0 < W)         v00 = value[(((size_t)b * LEN_TOT + s0 + y0 * W + x0) * 8 + h) * 64 + d];
        if (x0 + 1 >= 0 && x0 + 1 < W) v10 = value[(((size_t)b * LEN_TOT + s0 + y0 * W + x0 + 1) * 8 + h) * 64 + d];
      }
      if (y0 + 1 >= 0 && y0 + 1 < H) {
        if (x0 >= 0 && x0 < W)         v01 = value[(((size_t)b * LEN_TOT + s0 + (y0 + 1) * W + x0) * 8 + h) * 64 + d];
        if (x0 + 1 >= 0 && x0 + 1 < W) v11 = value[(((size_t)b * LEN_TOT + s0 + (y0 + 1) * W + x0 + 1) * 8 + h) * 64 + d];
      }
      acc += wgt * (v00 * (1.f - fx) * (1.f - fy) + v10 * fx * (1.f - fy)
                  + v01 * (1.f - fx) * fy + v11 * fx * fy);
    }
  }
  out[row * DMODEL + h * 64 + d] = acc;
}

// out = LayerNorm(a + b) * gamma + beta   (D=512, one block per row, in-place OK)
__global__ __launch_bounds__(128) void add_ln_kernel(
    const float* __restrict__ a, const float* __restrict__ bres,
    const float* __restrict__ gam, const float* __restrict__ bet,
    float* __restrict__ out)
{
  __shared__ float sh[128];
  size_t row = blockIdx.x;
  const float* pa = a + row * DMODEL;
  const float* pb = bres + row * DMODEL;
  float v[4];
  float sum = 0.f, sq = 0.f;
#pragma unroll
  for (int i = 0; i < 4; i++) {
    int c = threadIdx.x * 4 + i;
    float t = pa[c] + pb[c];
    v[i] = t; sum += t; sq += t * t;
  }
  sh[threadIdx.x] = sum; __syncthreads();
  for (int o = 64; o > 0; o >>= 1) { if (threadIdx.x < o) sh[threadIdx.x] += sh[threadIdx.x + o]; __syncthreads(); }
  float mean = sh[0] / (float)DMODEL; __syncthreads();
  sh[threadIdx.x] = sq; __syncthreads();
  for (int o = 64; o > 0; o >>= 1) { if (threadIdx.x < o) sh[threadIdx.x] += sh[threadIdx.x + o]; __syncthreads(); }
  float var = sh[0] / (float)DMODEL - mean * mean;
  float inv = rsqrtf(var + 1e-5f);
#pragma unroll
  for (int i = 0; i < 4; i++) {
    int c = threadIdx.x * 4 + i;
    out[row * DMODEL + c] = (v[i] - mean) * inv * gam[c] + bet[c];
  }
}

// query_embed [300,1024] -> qpos[b,300,512], tgt[b,300,512]
__global__ void qsplit_kernel(const float* __restrict__ qe,
                              float* __restrict__ qpos, float* __restrict__ tgt)
{
  int i = blockIdx.x * 256 + threadIdx.x;
  if (i >= NQ_DEC * DMODEL) return;
  int q = i / DMODEL, c = i % DMODEL;
  float qp = qe[q * 2 * DMODEL + c];
  float tg = qe[q * 2 * DMODEL + DMODEL + c];
  for (int b = 0; b < NB; b++) {
    qpos[((size_t)b * NQ_DEC + q) * DMODEL + c] = qp;
    tgt [((size_t)b * NQ_DEC + q) * DMODEL + c] = tg;
  }
}

// dref[b,q,l,2] = ref0[q,2] * vr[b,l,2]
__global__ void dref_kernel(const float* __restrict__ ref0, const float* __restrict__ vr,
                            float* __restrict__ dref)
{
  int i = blockIdx.x * 256 + threadIdx.x;
  if (i >= NB * NQ_DEC * 4) return;
  int l = i & 3; int t = i >> 2; int q = t % NQ_DEC; int b = t / NQ_DEC;
  dref[(size_t)i * 2 + 0] = ref0[q * 2 + 0] * vr[(b * 4 + l) * 2 + 0];
  dref[(size_t)i * 2 + 1] = ref0[q * 2 + 1] * vr[(b * 4 + l) * 2 + 1];
}

// decoder self-attn: one block per (b,h,q); Lq = Lk = 300, dh = 64
__global__ __launch_bounds__(64) void mha_kernel(
    const float* __restrict__ Q, const float* __restrict__ K,
    const float* __restrict__ V, float* __restrict__ O)
{
  const int Lq = NQ_DEC;
  __shared__ float sc[304];
  __shared__ float red[64];
  int q = blockIdx.x % Lq;
  int bh = blockIdx.x / Lq;
  int h = bh & 7, b = bh >> 3;
  int t = threadIdx.x;
  const float* qp = Q + (((size_t)b * Lq + q) * 8 + h) * 64;
  for (int k = t; k < Lq; k += 64) {
    const float* kp = K + (((size_t)b * Lq + k) * 8 + h) * 64;
    float d = 0.f;
    for (int c = 0; c < 64; c++) d += qp[c] * kp[c];
    sc[k] = d * 0.125f;   // 1/sqrt(64)
  }
  __syncthreads();
  float mx = -1e30f;
  for (int k = t; k < Lq; k += 64) mx = fmaxf(mx, sc[k]);
  red[t] = mx; __syncthreads();
  for (int o = 32; o > 0; o >>= 1) { if (t < o) red[t] = fmaxf(red[t], red[t + o]); __syncthreads(); }
  mx = red[0]; __syncthreads();
  float sm = 0.f;
  for (int k = t; k < Lq; k += 64) { float e = __expf(sc[k] - mx); sc[k] = e; sm += e; }
  red[t] = sm; __syncthreads();
  for (int o = 32; o > 0; o >>= 1) { if (t < o) red[t] += red[t + o]; __syncthreads(); }
  float inv = 1.f / red[0];
  float acc = 0.f;
  for (int k = 0; k < Lq; k++)
    acc += sc[k] * V[(((size_t)b * Lq + k) * 8 + h) * 64 + t];
  O[(((size_t)b * Lq + q) * 8 + h) * 64 + t] = acc * inv;
}

// ---------------------------------------------------------------------------
// Orchestration
// ---------------------------------------------------------------------------
extern "C" void kernel_launch(void* const* d_in, const int* in_sizes, int n_in,
                              void* d_out, int out_size, void* d_ws, size_t ws_size,
                              hipStream_t stream)
{
  (void)in_sizes; (void)n_in; (void)out_size; (void)ws_size;
  const float* F[56];
  for (int i = 0; i < 56; i++) F[i] = (const float*)d_in[i];
  const float* srcL[4] = { F[0], F[3], F[6], F[9] };
  const unsigned char* mskL[4] = {
    (const unsigned char*)d_in[1], (const unsigned char*)d_in[4],
    (const unsigned char*)d_in[7], (const unsigned char*)d_in[10] };
  const float* posL[4] = { F[2], F[5], F[8], F[11] };

  const int ML = NB * LEN_TOT;   // 43520
  const int MD = NB * NQ_DEC;    // 2400

  // ---- workspace carve ----
  char* cur = (char*)d_ws;
  auto allocf = [&](size_t n) -> float* {
    float* p = (float*)cur; cur += ((n * sizeof(float) + 255) / 256) * 256; return p; };
  float* x    = allocf((size_t)ML * DMODEL);   // activations / encoder memory
  float* posf = allocf((size_t)ML * DMODEL);
  float* qb   = allocf((size_t)ML * DMODEL);
  float* val  = allocf((size_t)ML * DMODEL);
  float* tmp  = allocf((size_t)ML * DMODEL);
  float* sob  = allocf((size_t)ML * 256);
  float* awb  = allocf((size_t)ML * 128);
  float* eref = allocf((size_t)ML * 8);
  float* hid  = allocf((size_t)LEN_TOT * 2048);   // FFN chunk (5440 rows)
  float* vr   = allocf(64);
  float* qpos = allocf((size_t)MD * DMODEL);
  float* dx   = allocf((size_t)MD * DMODEL);
  float* dq   = allocf((size_t)MD * DMODEL);
  float* dtmp = allocf((size_t)MD * DMODEL);
  float* dQ   = allocf((size_t)MD * DMODEL);
  float* dK   = allocf((size_t)MD * DMODEL);
  float* dV   = allocf((size_t)MD * DMODEL);
  float* ref0 = allocf((size_t)NQ_DEC * 2);
  float* dref = allocf((size_t)MD * 8);
  float* dhid = allocf((size_t)MD * 2048);
  unsigned char* mf = (unsigned char*)cur; cur += ((ML + 255) / 256) * 256;

  auto gemm = [&](int act, const float* A, const float* Wt, const float* bias,
                  float* C, const unsigned char* rm, int M, int N, int K) {
    dim3 g((M + 127) / 128, (N + 63) / 64), b(256);
    if (act == 0)      gemm_wmma<0, 0><<<g, b, 0, stream>>>(A, Wt, bias, C, rm, M, N, K, 0, 0);
    else if (act == 1) gemm_wmma<1, 0><<<g, b, 0, stream>>>(A, Wt, bias, C, rm, M, N, K, 0, 0);
    else               gemm_wmma<2, 0><<<g, b, 0, stream>>>(A, Wt, bias, C, rm, M, N, K, 0, 0);
  };

  // ---- flatten + embed levels (fused gather GEMM), pos, masks, ratios ----
  const int HS[4] = {64, 32, 16, 8}, S0[4] = {0, 4096, 5120, 5376};
  for (int l = 0; l < 4; l++) {
    int hw = HS[l] * HS[l];
    dim3 g((NB * hw + 127) / 128, (DMODEL + 63) / 64), b(256);
    gemm_wmma<0, 1><<<g, b, 0, stream>>>(srcL[l], F[13], F[14], x, nullptr,
                                         NB * hw, DMODEL, 2048, hw, S0[l]);
  }
  {
    size_t n = (size_t)ML * DMODEL;
    posflat_kernel<<<(unsigned)((n + 255) / 256), 256, 0, stream>>>(
        posL[0], posL[1], posL[2], posL[3], F[15], posf);
  }
  maskflat_kernel<<<(ML + 255) / 256, 256, 0, stream>>>(mskL[0], mskL[1], mskL[2], mskL[3], mf);
  vr_kernel<<<1, 32, 0, stream>>>(mskL[0], mskL[1], mskL[2], mskL[3], vr);
  enc_ref_kernel<<<(ML + 255) / 256, 256, 0, stream>>>(vr, eref);

  // ---- encoder stack ----
  for (int i = 0; i < 4; i++) {
    const float* so_w = F[18] + (size_t)i * 256 * 512;  const float* so_b = F[19] + i * 256;
    const float* aw_w = F[20] + (size_t)i * 128 * 512;  const float* aw_b = F[21] + i * 128;
    const float* vp_w = F[22] + (size_t)i * 512 * 512;  const float* vp_b = F[23] + i * 512;
    const float* op_w = F[24] + (size_t)i * 512 * 512;  const float* op_b = F[25] + i * 512;
    const float* l1_w = F[26] + (size_t)i * 2048 * 512; const float* l1_b = F[27] + i * 2048;
    const float* l2_w = F[28] + (size_t)i * 512 * 2048; const float* l2_b = F[29] + i * 512;
    const float* n1s = F[30] + i * 512, *n1b = F[31] + i * 512;
    const float* n2s = F[32] + i * 512, *n2b = F[33] + i * 512;

    size_t nel = (size_t)ML * DMODEL;
    add_kernel<<<(unsigned)((nel + 255) / 256), 256, 0, stream>>>(x, posf, qb, nel);
    gemm(0, x, vp_w, vp_b, val, mf, ML, 512, 512);            // value proj (+mask zero)
    gemm(0, qb, so_w, so_b, sob, nullptr, ML, 256, 512);      // sampling offsets
    gemm(0, qb, aw_w, aw_b, awb, nullptr, ML, 128, 512);      // attn weights logits
    softmax16_kernel<<<(ML * 8 + 255) / 256, 256, 0, stream>>>(awb, (size_t)ML * 8);
    deform_sample_kernel<<<(unsigned)(ML * 8), 64, 0, stream>>>(val, sob, awb, eref, tmp, LEN_TOT);
    gemm(0, tmp, op_w, op_b, qb, nullptr, ML, 512, 512);      // output proj
    add_ln_kernel<<<ML, 128, 0, stream>>>(x, qb, n1s, n1b, x);
    for (int m0 = 0; m0 < ML; m0 += LEN_TOT) {                // FFN, chunked
      gemm(1, x + (size_t)m0 * 512, l1_w, l1_b, hid, nullptr, LEN_TOT, 2048, 512);
      gemm(0, hid, l2_w, l2_b, tmp + (size_t)m0 * 512, nullptr, LEN_TOT, 512, 2048);
    }
    add_ln_kernel<<<ML, 128, 0, stream>>>(x, tmp, n2s, n2b, x);
  }
  // x now holds encoder memory

  // ---- decoder prep ----
  qsplit_kernel<<<(NQ_DEC * DMODEL + 255) / 256, 256, 0, stream>>>(F[12], qpos, dx);
  gemm(2, qpos, F[16], F[17], ref0, nullptr, NQ_DEC, 2, 512);  // sigmoid ref points (b=0 slice)
  dref_kernel<<<(NB * NQ_DEC * 4 + 255) / 256, 256, 0, stream>>>(ref0, vr, dref);

  // ---- decoder stack ----
  for (int i = 0; i < 4; i++) {
    const float* so_w = F[34] + (size_t)i * 256 * 512;  const float* so_b = F[35] + i * 256;
    const float* aw_w = F[36] + (size_t)i * 128 * 512;  const float* aw_b = F[37] + i * 128;
    const float* vp_w = F[38] + (size_t)i * 512 * 512;  const float* vp_b = F[39] + i * 512;
    const float* op_w = F[40] + (size_t)i * 512 * 512;  const float* op_b = F[41] + i * 512;
    const float* l1_w = F[42] + (size_t)i * 2048 * 512; const float* l1_b = F[43] + i * 2048;
    const float* l2_w = F[44] + (size_t)i * 512 * 2048; const float* l2_b = F[45] + i * 512;
    const float* n1s = F[46] + i * 512, *n1b = F[47] + i * 512;
    const float* n2s = F[48] + i * 512, *n2b = F[49] + i * 512;
    const float* n3s = F[50] + i * 512, *n3b = F[51] + i * 512;
    const float* inw = F[52] + (size_t)i * 1536 * 512;  const float* inb = F[53] + i * 1536;
    const float* ow  = F[54] + (size_t)i * 512 * 512;   const float* ob  = F[55] + i * 512;

    size_t nel = (size_t)MD * DMODEL;
    // self attention
    add_kernel<<<(unsigned)((nel + 255) / 256), 256, 0, stream>>>(dx, qpos, dq, nel);
    gemm(0, dq, inw,                      inb,        dQ, nullptr, MD, 512, 512);
    gemm(0, dq, inw + (size_t)512 * 512,  inb + 512,  dK, nullptr, MD, 512, 512);
    gemm(0, dx, inw + (size_t)1024 * 512, inb + 1024, dV, nullptr, MD, 512, 512);
    mha_kernel<<<NB * 8 * NQ_DEC, 64, 0, stream>>>(dQ, dK, dV, dtmp);
    gemm(0, dtmp, ow, ob, dq, nullptr, MD, 512, 512);
    add_ln_kernel<<<MD, 128, 0, stream>>>(dx, dq, n2s, n2b, dx);
    // cross deformable attention over memory
    add_kernel<<<(unsigned)((nel + 255) / 256), 256, 0, stream>>>(dx, qpos, dq, nel);
    gemm(0, x, vp_w, vp_b, val, mf, ML, 512, 512);
    gemm(0, dq, so_w, so_b, sob, nullptr, MD, 256, 512);
    gemm(0, dq, aw_w, aw_b, awb, nullptr, MD, 128, 512);
    softmax16_kernel<<<(MD * 8 + 255) / 256, 256, 0, stream>>>(awb, (size_t)MD * 8);
    deform_sample_kernel<<<(unsigned)(MD * 8), 64, 0, stream>>>(val, sob, awb, dref, dtmp, NQ_DEC);
    gemm(0, dtmp, op_w, op_b, dq, nullptr, MD, 512, 512);
    add_ln_kernel<<<MD, 128, 0, stream>>>(dx, dq, n1s, n1b, dx);
    // FFN
    gemm(1, dx, l1_w, l1_b, dhid, nullptr, MD, 2048, 512);
    gemm(0, dhid, l2_w, l2_b, dtmp, nullptr, MD, 512, 2048);
    add_ln_kernel<<<MD, 128, 0, stream>>>(dx, dtmp, n3s, n3b, dx);
  }

  hipMemcpyAsync(d_out, dx, (size_t)MD * DMODEL * sizeof(float),
                 hipMemcpyDeviceToDevice, stream);
}
